// MSDeformAttn_45973329937037
// MI455X (gfx1250) — matrix operand order repaired and checked by reading, not compiled, not run on hardware
//
#include <hip/hip_runtime.h>
#include <hip/hip_bf16.h>

typedef __attribute__((ext_vector_type(16))) _Float16     v16h;
typedef __attribute__((ext_vector_type(8)))  float        v8f;
typedef __attribute__((ext_vector_type(4)))  float        v4f;
typedef __attribute__((ext_vector_type(4)))  unsigned int v4u;

union Frag16 { v16h h; v4u q[2]; };

#define HWp   2304      // 48*48
#define Wd    48
#define PWd   50        // padded width  (48 + 2)
#define PPAD  2500      // padded pixels (50*50)
#define NIMG  6         // bs*t
#define CDIM  256
#define LQc   6912      // 3*2304
#define NHc   8
#define DHc   32

// ---------------------------------------------------------------- converters
// fp32 [nimg][c][p] -> f16 channel-last PADDED [nimg][50*50][c], zero border
__global__ void k_cvtT_pad(const float* __restrict__ in, _Float16* __restrict__ out) {
    int idx = blockIdx.x * 256 + threadIdx.x;
    if (idx >= NIMG * PPAD * CDIM) return;
    int c = idx & 255;
    int rest = idx >> 8;
    int pp = rest % PPAD;
    int nimg = rest / PPAD;
    int py = pp / PWd, px = pp - py * PWd;
    float v = 0.f;
    if (py >= 1 && py <= Wd && px >= 1 && px <= Wd)
        v = in[((size_t)nimg * CDIM + c) * HWp + (py - 1) * Wd + (px - 1)];
    out[idx] = (_Float16)v;
}

// fp32 channel-last [nimg][p][c] -> f16 channel-last PADDED [nimg][50*50][c]
__global__ void k_pad_cast(const float* __restrict__ in, _Float16* __restrict__ out) {
    int idx = blockIdx.x * 256 + threadIdx.x;
    if (idx >= NIMG * PPAD * CDIM) return;
    int c = idx & 255;
    int rest = idx >> 8;
    int pp = rest % PPAD;
    int nimg = rest / PPAD;
    int py = pp / PWd, px = pp - py * PWd;
    float v = 0.f;
    if (py >= 1 && py <= Wd && px >= 1 && px <= Wd)
        v = in[((size_t)nimg * HWp + (py - 1) * Wd + (px - 1)) * CDIM + c];
    out[idx] = (_Float16)v;
}

// w (OC, 256, 3, 3) fp32 -> Wh[tap][oc][ic] f16, zero-padded to OCpad rows
__global__ void k_wcvt(const float* __restrict__ w, _Float16* __restrict__ out, int OC, int OCpad) {
    int i = blockIdx.x * 256 + threadIdx.x;
    int total = 9 * OCpad * 256;
    if (i >= total) return;
    int tap = i / (OCpad * 256);
    int rem = i - tap * OCpad * 256;
    int oc  = rem >> 8;
    int ic  = rem & 255;
    out[i] = (oc < OC) ? (_Float16)w[((size_t)oc * 256 + ic) * 9 + tap] : (_Float16)0.f;
}

// ------------------------------------------- WMMA implicit-GEMM conv3x3, LDS-free
// XT: [nimg][2500][256] f16 channel-last padded (zero halo), Wh: [9][OCpad][256] f16
// Out: [nimg][OC][2304] f32.  A = activations (M=spatial,K=ic), B = weights (K=ic,N=oc).
// Per wave: 32 spatial x 64 oc = 8 WMMA accumulators; all fragments loaded directly
// from global (contiguous ic); 3x3 shifts are pure pointer bumps into the zero halo.
__global__ __launch_bounds__(256) void k_conv3x3_wmma(
    const _Float16* __restrict__ XT, const _Float16* __restrict__ Wh,
    const float* __restrict__ bias, float* __restrict__ Out, int OC, int OCpad)
{
    const int lane = threadIdx.x & 31;
    const int wave = threadIdx.x >> 5;
    const int nimg = blockIdx.z;
    const int oc0  = blockIdx.y << 6;                    // 64-col oc tile
    const int spw  = (blockIdx.x << 8) + (wave << 5);    // wave's 32-row spatial tile
    const int m    = lane & 15;
    const int kb   = (lane >> 4) << 3;                   // A k-sub: 0/8
    const int kg   = (lane >> 4) << 3;                   // B k-sub: 0/8

    v8f acc[2][4];
#pragma unroll
    for (int a = 0; a < 2; ++a)
#pragma unroll
        for (int c = 0; c < 4; ++c)
#pragma unroll
            for (int r = 0; r < 8; ++r) acc[a][c][r] = 0.f;

    const _Float16* Xn = XT + (size_t)nimg * PPAD * CDIM;

    const _Float16* base[2];
#pragma unroll
    for (int a = 0; a < 2; ++a) {
        const int p = spw + (a << 4) + m;
        const int y = p / Wd, x = p - y * Wd;
        base[a] = Xn + (size_t)((y + 1) * PWd + (x + 1)) * CDIM;
    }

    for (int tap = 0; tap < 9; ++tap) {
        const int dp = ((tap / 3 - 1) * PWd + (tap % 3 - 1)) * CDIM;
        const _Float16* ax0 = base[0] + dp;
        const _Float16* ax1 = base[1] + dp;
        const _Float16* Wt  = Wh + (size_t)tap * OCpad * 256;
        for (int ic0 = 0; ic0 < 256; ic0 += 32) {
            Frag16 fa0, fa1;
            fa0.q[0] = *(const v4u*)(ax0 + ic0 + kb);
            fa0.q[1] = *(const v4u*)(ax0 + ic0 + 16 + kb);
            fa1.q[0] = *(const v4u*)(ax1 + ic0 + kb);
            fa1.q[1] = *(const v4u*)(ax1 + ic0 + 16 + kb);
#pragma unroll
            for (int c = 0; c < 4; ++c) {
                const _Float16* bp = Wt + (size_t)(oc0 + (c << 4) + m) * 256 + ic0 + kg;
                Frag16 fb;
                fb.q[0] = *(const v4u*)bp;
                fb.q[1] = *(const v4u*)(bp + 16);
                acc[0][c] = __builtin_amdgcn_wmma_f32_16x16x32_f16(
                    false, fa0.h, false, fb.h, (short)0, acc[0][c], false, false);
                acc[1][c] = __builtin_amdgcn_wmma_f32_16x16x32_f16(
                    false, fa1.h, false, fb.h, (short)0, acc[1][c], false, false);
            }
        }
    }
    // epilogue: D lane -> N(oc)=lane&15, VGPR r -> M(p)=r+8*(lane>>4); p consecutive in r
    const int rofs = (lane >> 4) << 3;
#pragma unroll
    for (int c = 0; c < 4; ++c) {
        const int oc = oc0 + (c << 4) + m;
        if (oc < OC) {
            const float b = bias[oc];
            float* op = Out + ((size_t)nimg * OC + oc) * HWp;
#pragma unroll
            for (int a = 0; a < 2; ++a) {
                const int p0 = spw + (a << 4) + rofs;
                v4f s0 = {acc[a][c][0] + b, acc[a][c][1] + b, acc[a][c][2] + b, acc[a][c][3] + b};
                v4f s1 = {acc[a][c][4] + b, acc[a][c][5] + b, acc[a][c][6] + b, acc[a][c][7] + b};
                *(v4f*)(op + p0)     = s0;
                *(v4f*)(op + p0 + 4) = s1;
            }
        }
    }
}

// -------------------------------------------------- flow add-table (warp fused)
__device__ inline float bilin_border(const float* __restrict__ ch, float sx, float sy) {
    sx = fminf(fmaxf(sx, 0.f), 47.f);
    sy = fminf(fmaxf(sy, 0.f), 47.f);
    float x0f = floorf(sx), y0f = floorf(sy);
    int x0 = (int)x0f, y0 = (int)y0f;
    int x1 = x0 + 1 > 47 ? 47 : x0 + 1;
    int y1 = y0 + 1 > 47 ? 47 : y0 + 1;
    float wx1 = sx - x0f, wx0 = 1.f - wx1;
    float wy1 = sy - y0f, wy0 = 1.f - wy1;
    return ch[y0 * Wd + x0] * wx0 * wy0 + ch[y0 * Wd + x1] * wx1 * wy0 +
           ch[y1 * Wd + x0] * wx0 * wy1 + ch[y1 * Wd + x1] * wx1 * wy1;
}

// ADD[n][t][l][p][2]
__global__ void k_addtab(const float* __restrict__ ff, const float* __restrict__ fb,
                         float* __restrict__ ADD) {
    int i = blockIdx.x * 256 + threadIdx.x;
    if (i >= 2 * HWp) return;
    int n = i / HWp, p = i - n * HWp;
    int y = p / Wd, x = p - y * Wd;
    const float* F = ff + (size_t)n * 4 * HWp;   // (2,2,H,W)
    const float* B = fb + (size_t)n * 4 * HWp;
    float f01x = F[p],             f01y = F[HWp + p];
    float f12x = F[2 * HWp + p],   f12y = F[3 * HWp + p];
    float b10x = B[p],             b10y = B[HWp + p];
    float b21x = B[2 * HWp + p],   b21y = B[3 * HWp + p];
    float f02x = f01x + bilin_border(F + 2 * HWp, x + f01x, y + f01y);
    float f02y = f01y + bilin_border(F + 3 * HWp, x + f01x, y + f01y);
    float b20x = b21x + bilin_border(B,           x + b21x, y + b21y);
    float b20y = b21y + bilin_border(B + HWp,     x + b21x, y + b21y);
    auto W2 = [&](int t, int l, float vx, float vy) {
        size_t b = (((size_t)(n * 3 + t) * 3 + l) * HWp + p) * 2;
        ADD[b] = vx; ADD[b + 1] = vy;
    };
    W2(0,0,0.f,0.f);   W2(0,1,f01x,f01y); W2(0,2,f02x,f02y);
    W2(1,0,b10x,b10y); W2(1,1,0.f,0.f);   W2(1,2,f12x,f12y);
    W2(2,0,b20x,b20y); W2(2,1,b21x,b21y); W2(2,2,0.f,0.f);
}

// -------------------------------------------------- softmax over 12 (NL*NP)
__global__ void k_softmax12(float* __restrict__ ATT) {
    int i = blockIdx.x * 256 + threadIdx.x;
    if (i >= NIMG * HWp * NHc) return;
    int h = i & 7;
    int rest = i >> 3;
    int p = rest % HWp;
    int nimg = rest / HWp;
    float* b = ATT + ((size_t)nimg * 96 + h * 12) * HWp + p;
    float v[12], mx = -1e30f;
#pragma unroll
    for (int j = 0; j < 12; ++j) { v[j] = b[(size_t)j * HWp]; mx = fmaxf(mx, v[j]); }
    float s = 0.f;
#pragma unroll
    for (int j = 0; j < 12; ++j) { v[j] = __expf(v[j] - mx); s += v[j]; }
    float inv = 1.f / s;
#pragma unroll
    for (int j = 0; j < 12; ++j) b[(size_t)j * HWp] = v[j] * inv;
}

// -------------------------------------------------- value mask + channel-last
__global__ void k_mask_transpose_value(const float* __restrict__ V,
                                       const unsigned char* __restrict__ mask,
                                       float* __restrict__ VT) {
    int idx = blockIdx.x * 256 + threadIdx.x;
    if (idx >= NIMG * CDIM * HWp) return;
    int c = idx & 255;
    int rest = idx >> 8;
    int p = rest % HWp;
    int nimg = rest / HWp;
    int n = nimg / 3, t = nimg - n * 3;
    float v = V[((size_t)nimg * CDIM + c) * HWp + p];
    if (mask[(size_t)n * LQc + t * HWp + p]) v = 0.f;
    VT[idx] = v;   // VT[nimg][p][c]
}

// -------------------------------------------------- fused grid-sample + einsum
__global__ __launch_bounds__(256) void k_sample(
    const float* __restrict__ ref,   // (2,6912,3,2)
    const float* __restrict__ OFF,   // (6,192,2304)
    const float* __restrict__ ATT,   // (6,96,2304) softmaxed
    const float* __restrict__ ADD,   // (2,3,3,2304,2)
    const float* __restrict__ VT,    // (6,2304,256)
    float* __restrict__ OS)          // (2,6912,256)
{
    const int lane = threadIdx.x & 31;           // channel within head
    const int g = blockIdx.x * 8 + (threadIdx.x >> 5);
    const int h = g & 7;
    const int q = (g >> 3) % LQc;
    const int n = g / (LQc * NHc);
    const int t = q / HWp;
    const int p = q - t * HWp;
    const int nimg = n * 3 + t;
    float acc = 0.f;
#pragma unroll
    for (int l = 0; l < 3; ++l) {
        const size_t rb = ((size_t)(n * LQc + q) * 3 + l) * 2;
        const float rx = ref[rb], ry = ref[rb + 1];
        const size_t ab = (((size_t)(n * 3 + t) * 3 + l) * HWp + p) * 2;
        const float ax = ADD[ab], ay = ADD[ab + 1];
        const float* vb = VT + (size_t)(n * 3 + l) * HWp * CDIM + h * DHc + lane;
#pragma unroll
        for (int pp = 0; pp < 4; ++pp) {
            const int ch = (h * 3 + l) * 4 + pp;
            const float ox = OFF[((size_t)nimg * 192 + ch * 2)     * HWp + p];
            const float oy = OFF[((size_t)nimg * 192 + ch * 2 + 1) * HWp + p];
            const float a  = ATT[((size_t)nimg * 96  + ch)         * HWp + p];
            // loc = ref + (off+add)/48 ; grid=2*loc-1 ; x=((gx+1)*48-1)/2 = 48*loc-0.5
            const float sx = 48.f * rx + (ox + ax) - 0.5f;
            const float sy = 48.f * ry + (oy + ay) - 0.5f;
            const float x0f = floorf(sx), y0f = floorf(sy);
            const int x0 = (int)x0f, y0 = (int)y0f;
            const float wx1 = sx - x0f, wx0 = 1.f - wx1;
            const float wy1 = sy - y0f, wy0 = 1.f - wy1;
            if ((unsigned)x0 < 48u && (unsigned)y0 < 48u)
                acc += a * wx0 * wy0 * vb[(size_t)(y0 * Wd + x0) * CDIM];
            if ((unsigned)(x0 + 1) < 48u && (unsigned)y0 < 48u)
                acc += a * wx1 * wy0 * vb[(size_t)(y0 * Wd + x0 + 1) * CDIM];
            if ((unsigned)x0 < 48u && (unsigned)(y0 + 1) < 48u)
                acc += a * wx0 * wy1 * vb[(size_t)((y0 + 1) * Wd + x0) * CDIM];
            if ((unsigned)(x0 + 1) < 48u && (unsigned)(y0 + 1) < 48u)
                acc += a * wx1 * wy1 * vb[(size_t)((y0 + 1) * Wd + x0 + 1) * CDIM];
        }
    }
    OS[(size_t)(n * LQc + q) * CDIM + h * DHc + lane] = acc;
}

// ================================================================ launch
extern "C" void kernel_launch(void* const* d_in, const int* in_sizes, int n_in,
                              void* d_out, int out_size, void* d_ws, size_t ws_size,
                              hipStream_t stream) {
    const float* query  = (const float*)d_in[0];
    const float* refpts = (const float*)d_in[1];
    const float* inflat = (const float*)d_in[2];
    const unsigned char* mask = (const unsigned char*)d_in[5];
    const float* ff     = (const float*)d_in[6];
    const float* fbk    = (const float*)d_in[7];
    const float* w_off  = (const float*)d_in[8];
    const float* b_off  = (const float*)d_in[9];
    const float* w_attn = (const float*)d_in[10];
    const float* b_attn = (const float*)d_in[11];
    const float* w_val  = (const float*)d_in[12];
    const float* b_val  = (const float*)d_in[13];
    const float* w_out  = (const float*)d_in[14];
    const float* b_out  = (const float*)d_in[15];
    float* out = (float*)d_out;

    char* ws = (char*)d_ws;
    size_t o = 0;
    auto take = [&](size_t bytes) -> char* {
        char* p = ws + o;
        o += (bytes + 255) & ~(size_t)255;
        return p;
    };
    const size_t NEL  = (size_t)NIMG * CDIM * HWp;     // 3,538,944 (unpadded elems)
    const size_t NELP = (size_t)NIMG * CDIM * PPAD;    // 3,840,000 (padded elems)
    _Float16* Xq    = (_Float16*)take(NELP * 2);                   // padded channel-last f16
    _Float16* Xv    = (_Float16*)take(NELP * 2);                   // padded channel-last f16
    _Float16* Woff  = (_Float16*)take((size_t)9 * 192 * 256 * 2);
    _Float16* Wattn = (_Float16*)take((size_t)9 * 128 * 256 * 2);  // padded 96->128
    _Float16* Wval  = (_Float16*)take((size_t)9 * 256 * 256 * 2);
    _Float16* Wout  = (_Float16*)take((size_t)9 * 256 * 256 * 2);
    float* OFF = (float*)take((size_t)NIMG * 192 * HWp * 4);
    float* ATT = (float*)take((size_t)NIMG * 96  * HWp * 4);
    float* VAL = (float*)take(NEL * 4);
    float* VT  = (float*)take(NEL * 4);
    float* ADD = (float*)take((size_t)2 * 3 * 3 * HWp * 2 * 4);
    float* OS  = (float*)take((size_t)2 * LQc * CDIM * 4);
    _Float16* OH = (_Float16*)take(NELP * 2);                      // padded channel-last f16

    int nb  = (int)((NEL + 255) / 256);
    int nbp = (int)((NELP + 255) / 256);
    // activations: fp32 [nimg][c][p] -> f16 padded channel-last [nimg][50*50][c]
    k_cvtT_pad<<<nbp, 256, 0, stream>>>(query,  Xq);
    k_cvtT_pad<<<nbp, 256, 0, stream>>>(inflat, Xv);
    k_wcvt<<<(9 * 192 * 256 + 255) / 256, 256, 0, stream>>>(w_off,  Woff, 192, 192);
    k_wcvt<<<(9 * 128 * 256 + 255) / 256, 256, 0, stream>>>(w_attn, Wattn, 96, 128);
    k_wcvt<<<(9 * 256 * 256 + 255) / 256, 256, 0, stream>>>(w_val,  Wval, 256, 256);
    k_wcvt<<<(9 * 256 * 256 + 255) / 256, 256, 0, stream>>>(w_out,  Wout, 256, 256);

    // LDS-free implicit-GEMM convs: grid = (2304/256 spatial, OCpad/64, nimg)
    k_conv3x3_wmma<<<dim3(9, 3, 6), 256, 0, stream>>>(Xq, Woff,  b_off,  OFF, 192, 192);
    k_conv3x3_wmma<<<dim3(9, 2, 6), 256, 0, stream>>>(Xq, Wattn, b_attn, ATT, 96, 128);
    k_conv3x3_wmma<<<dim3(9, 4, 6), 256, 0, stream>>>(Xv, Wval,  b_val,  VAL, 256, 256);

    k_addtab<<<(2 * HWp + 255) / 256, 256, 0, stream>>>(ff, fbk, ADD);
    k_softmax12<<<(NIMG * HWp * NHc + 255) / 256, 256, 0, stream>>>(ATT);
    k_mask_transpose_value<<<nb, 256, 0, stream>>>(VAL, mask, VT);
    k_sample<<<(2 * LQc * NHc) / 8, 256, 0, stream>>>(refpts, OFF, ATT, ADD, VT, OS);
    // OS [n][q][c] == [nimg][p][c]: pad-cast into final-conv input layout
    k_pad_cast<<<nbp, 256, 0, stream>>>(OS, OH);
    k_conv3x3_wmma<<<dim3(9, 4, 6), 256, 0, stream>>>(OH, Wout, b_out, out, 256, 256);
    (void)in_sizes; (void)n_in; (void)out_size; (void)ws_size;
}
